// CQTExtractor_5738076307832
// MI455X (gfx1250) — matrix operand order repaired and checked by reading, not compiled, not run on hardware
//
#include <hip/hip_runtime.h>
#include <hip/hip_bf16.h>
#include <math.h>

// ---------------------------------------------------------------------------
// CQT extractor for gfx1250 (MI455X).
// DFT-as-WMMA: spectrum = frames x DFT(cos/sin) via v_wmma_f32_16x16x32_f16,
// magnitude in LDS (f16), then CQT projection as a second WMMA pass.
// 48 frames/block (3 M-tiles share each B load) -> 7.3 GB L2 table traffic,
// 6 WMMAs per table fetch, ~300 KB LDS (within the 320 KB WGP budget).
// ---------------------------------------------------------------------------

typedef __attribute__((ext_vector_type(16))) _Float16 v16h;
typedef __attribute__((ext_vector_type(8)))  _Float16 h8;
typedef __attribute__((ext_vector_type(8)))  float    v8f;

#define AUDIO_LEN   1310720
#define NFRAMES     2561
#define NFFT        2048
#define HOP         512
#define NBINS       84

#define FPB         48          // frames per block (three 16-row M tiles)
#define T_TILES     54          // ceil(2561 / 48)

#define FT_TILES    65          // freq tiles of 16 -> covers 1040 >= 1025
#define KCH_DFT     64          // 2048 / 32 K-chunks for the DFT pass
#define MT_TILES    6           // bin tiles of 16 -> 96 >= 84
#define KCH_PRJ     33          // 1056 / 32 K-chunks for the projection pass

#define SA_STRIDE   2056        // halves; 4112 B rows: 16B aligned, banks spread
#define MAG_STRIDE  1064        // halves; 2128 B rows: 16B aligned, banks spread

#define DFT_ELEMS   (FT_TILES * KCH_DFT * 32 * 16)   // 2,129,920 halves / table
#define W_ELEMS     (MT_TILES * KCH_PRJ * 32 * 16)   // 101,376 halves

// WGP-scope prefetch (scope 0 = "pull into all cache levels on miss").
// __builtin_prefetch can only reach SE/DEV/SYS scopes, which skip the WGP$.
__device__ __forceinline__ void prefetch_wgp(const void* p) {
    asm volatile("global_prefetch_b8 %0, off" : : "v"(p));
}

// ---------------------------------------------------------------------------
// DFT cos/sin tables, pre-swizzled into a contiguous-K WMMA B per-lane order (matches A loads)
// (16-bit B, 32x16): lane l holds N = l%16, elements e -> K = (l<16?0:16)+e.
// Linear index = ((ftile*64 + kchunk)*32 + lane)*16 + e  ->  32B/lane coalesced.
// Exact twiddle reduction: theta = ((f*j) mod 2048) * pi/1024.
// ---------------------------------------------------------------------------
__global__ __launch_bounds__(256) void cqt_dft_tables(_Float16* __restrict__ dcos,
                                                      _Float16* __restrict__ dsin) {
    int idx = blockIdx.x * 256 + threadIdx.x;
    if (idx >= DFT_ELEMS) return;
    int e    = idx & 15;
    int lane = (idx >> 4) & 31;
    int kc   = (idx >> 9) & 63;
    int ft   = idx >> 15;
    int f = ft * 16 + (lane & 15);
    int j = kc * 32 + ((lane < 16) ? 0 : 16) + e;
    float c = 0.f, s = 0.f;
    if (f <= 1024) {                       // 1025 rfft bins; pad rows are zero
        int r = (f * j) & (NFFT - 1);
        float th = (float)r * 3.0679615757712823e-3f;   // pi/1024
        __sincosf(th, &s, &c);
    }
    dcos[idx] = (_Float16)c;
    dsin[idx] = (_Float16)s;               // sign of im irrelevant under |.|
}

// ---------------------------------------------------------------------------
// CQT weights, pre-swizzled into the WMMA A-matrix per-lane layout
// (16-bit A, 16x32): lane l holds M = l%16; elems 0..7 -> K{0..7} (lo) / {8..15}
// (hi); elems 8..15 -> K{16..23} (lo) / {24..31} (hi).
// ---------------------------------------------------------------------------
__global__ __launch_bounds__(256) void cqt_weights(_Float16* __restrict__ wswz) {
    int idx = blockIdx.x * 256 + threadIdx.x;
    if (idx >= W_ELEMS) return;
    int e    = idx & 15;
    int lane = (idx >> 4) & 31;
    int lin  = idx >> 9;                   // (mt * 33 + kc)
    int kc   = lin % KCH_PRJ;
    int mt   = lin / KCH_PRJ;
    int bin  = mt * 16 + (lane & 15);
    int koff = (lane < 16) ? ((e < 8) ? e : e + 8)
                           : ((e < 8) ? e + 8 : e + 16);
    int f = kc * 32 + koff;
    float v = 0.f;
    if (bin < NBINS && f <= 1024) {
        float cq = 27.5f * exp2f((float)bin * (1.0f / 12.0f));
        float fr = (float)f * (22050.0f / 2048.0f);
        v = __expf(-fabsf(fr - cq) / (0.1f * cq));
    }
    wswz[idx] = (_Float16)v;
}

// One DFT K-step: 3 A tiles (frame thirds) from LDS, 6 WMMAs vs shared bc/bs.
__device__ __forceinline__ void dft_step(const _Float16* __restrict__ rowA0,
                                         const _Float16* __restrict__ rowA1,
                                         const _Float16* __restrict__ rowA2,
                                         int kc, bool lo, v16h bc, v16h bs,
                                         v8f& cre0, v8f& cim0,
                                         v8f& cre1, v8f& cim1,
                                         v8f& cre2, v8f& cim2) {
    union { v16h v; h8 h[2]; } a0, a1, a2;
    const int o0 = kc * 32 + (lo ? 0 : 16);
    const int o1 = kc * 32 + (lo ? 8 : 24);
    a0.h[0] = *(const h8*)(rowA0 + o0);
    a0.h[1] = *(const h8*)(rowA0 + o1);
    a1.h[0] = *(const h8*)(rowA1 + o0);
    a1.h[1] = *(const h8*)(rowA1 + o1);
    a2.h[0] = *(const h8*)(rowA2 + o0);
    a2.h[1] = *(const h8*)(rowA2 + o1);
    cre0 = __builtin_amdgcn_wmma_f32_16x16x32_f16(false, a0.v, false, bc, (short)0, cre0, false, false);
    cim0 = __builtin_amdgcn_wmma_f32_16x16x32_f16(false, a0.v, false, bs, (short)0, cim0, false, false);
    cre1 = __builtin_amdgcn_wmma_f32_16x16x32_f16(false, a1.v, false, bc, (short)0, cre1, false, false);
    cim1 = __builtin_amdgcn_wmma_f32_16x16x32_f16(false, a1.v, false, bs, (short)0, cim1, false, false);
    cre2 = __builtin_amdgcn_wmma_f32_16x16x32_f16(false, a2.v, false, bc, (short)0, cre2, false, false);
    cim2 = __builtin_amdgcn_wmma_f32_16x16x32_f16(false, a2.v, false, bs, (short)0, cim2, false, false);
}

// ---------------------------------------------------------------------------
// Main kernel: one block (256 threads = 8 wave32) handles 48 frames.
//  Phase 0: window + reflect-pad frames -> LDS f16 (48 x 2048).
//  Phase A: DFT via WMMA (cos & sin), |.| -> LDS mag (f16, 48 x 1056, padded 0).
//  Phase B: 96x48 CQT projection via WMMA from LDS, log10 -> out.
// ---------------------------------------------------------------------------
__global__ __launch_bounds__(256) void cqt_main(const float*    __restrict__ audio,
                                                const _Float16* __restrict__ dcos,
                                                const _Float16* __restrict__ dsin,
                                                const _Float16* __restrict__ wswz,
                                                float*          __restrict__ out) {
    extern __shared__ _Float16 smem[];
    _Float16* sA = smem;                        // FPB x SA_STRIDE  (windowed frames)
    _Float16* sM = smem + FPB * SA_STRIDE;      // FPB x MAG_STRIDE (magnitudes)

    const int tile = blockIdx.x;                // frame tile (54 tiles of 48)
    const int b    = blockIdx.y;                // batch
    const int tid  = threadIdx.x;
    const int lane = tid & 31;
    const int wave = tid >> 5;
    const int t16  = lane & 15;
    const bool lo  = (lane < 16);

    // Zero the magnitude buffer (projection K is padded to 1056).
    for (int i = tid; i < FPB * MAG_STRIDE; i += 256) sM[i] = (_Float16)0.f;

    // Phase 0: gather frames with reflect padding + Hann window -> f16 LDS.
    const float* aud = audio + (size_t)b * AUDIO_LEN;
    for (int s = tid; s < FPB * NFFT; s += 256) {
        int fr = s >> 11;
        int j  = s & (NFFT - 1);
        int i  = (tile * FPB + fr) * HOP + j - (NFFT / 2);
        if (i < 0) i = -i;                                   // reflect left
        if (i >= AUDIO_LEN) i = 2 * (AUDIO_LEN - 1) - i;     // reflect right
        float w = 0.5f - 0.5f * __cosf((float)j * (6.283185307179586f / 2048.0f));
        sA[fr * SA_STRIDE + j] = (_Float16)(aud[i] * w);
    }
    __syncthreads();

    // Phase A: spectrum magnitudes. Each wave owns freq tiles nt = wave, wave+8, ...
    for (int nt = wave; nt < FT_TILES; nt += 8) {
        v8f cre0 = {}, cim0 = {}, cre1 = {}, cim1 = {}, cre2 = {}, cim2 = {};
        const v16h* pc = ((const v16h*)dcos) + (size_t)nt * KCH_DFT * 32 + lane;
        const v16h* ps = ((const v16h*)dsin) + (size_t)nt * KCH_DFT * 32 + lane;
        const _Float16* rowA0 = sA + t16 * SA_STRIDE;          // frames 0..15
        const _Float16* rowA1 = sA + (t16 + 16) * SA_STRIDE;   // frames 16..31
        const _Float16* rowA2 = sA + (t16 + 32) * SA_STRIDE;   // frames 32..47

        // Double-buffered B stream; WGP-scope prefetch 8 chunks (8 KB) ahead.
        v16h bc = pc[0];
        v16h bs = ps[0];
        int kc = 0;
        for (; kc < KCH_DFT - 8; ++kc) {                       // body with prefetch
            v16h bcn = pc[(size_t)(kc + 1) * 32];
            v16h bsn = ps[(size_t)(kc + 1) * 32];
            prefetch_wgp((const void*)(pc + (size_t)(kc + 8) * 32));
            prefetch_wgp((const void*)(ps + (size_t)(kc + 8) * 32));
            dft_step(rowA0, rowA1, rowA2, kc, lo, bc, bs,
                     cre0, cim0, cre1, cim1, cre2, cim2);
            bc = bcn; bs = bsn;
        }
        for (; kc < KCH_DFT - 1; ++kc) {                       // tail, no prefetch
            v16h bcn = pc[(size_t)(kc + 1) * 32];
            v16h bsn = ps[(size_t)(kc + 1) * 32];
            dft_step(rowA0, rowA1, rowA2, kc, lo, bc, bs,
                     cre0, cim0, cre1, cim1, cre2, cim2);
            bc = bcn; bs = bsn;
        }
        dft_step(rowA0, rowA1, rowA2, KCH_DFT - 1, lo, bc, bs,
                 cre0, cim0, cre1, cim1, cre2, cim2);

        // C/D layout: VGPR r -> M = r (lanes 0-15) / r+8 (lanes 16-31), N = lane%16.
        int f = nt * 16 + t16;
#pragma unroll
        for (int r = 0; r < 8; ++r) {
            int tt = lo ? r : (r + 8);
            float m0 = sqrtf(cre0[r] * cre0[r] + cim0[r] * cim0[r]);
            float m1 = sqrtf(cre1[r] * cre1[r] + cim1[r] * cim1[r]);
            float m2 = sqrtf(cre2[r] * cre2[r] + cim2[r] * cim2[r]);
            sM[tt * MAG_STRIDE + f]        = (_Float16)m0;
            sM[(tt + 16) * MAG_STRIDE + f] = (_Float16)m1;
            sM[(tt + 32) * MAG_STRIDE + f] = (_Float16)m2;
        }
    }
    __syncthreads();

    // Phase B: cqt[bin, t] = sum_f w[bin,f] * mag[t,f].
    // 18 tasks = 6 bin-tiles x 3 frame thirds, distributed over 8 waves.
    for (int task = wave; task < MT_TILES * 3; task += 8) {
        const int mt = task % MT_TILES;       // bin tile
        const int nh = task / MT_TILES;       // frame third (t offset nh*16)
        v8f acc = {};
        const v16h* pw = ((const v16h*)wswz) + (size_t)mt * KCH_PRJ * 32 + lane;
        const _Float16* rowM = sM + (t16 + nh * 16) * MAG_STRIDE + (lo ? 0 : 8);
        for (int kc = 0; kc < KCH_PRJ; ++kc) {
            union { v16h v; h8 h[2]; } bm;    // B: N = frame (lane%16), K = freq
            bm.h[0] = *(const h8*)(rowM + kc * 32);
            bm.h[1] = *(const h8*)(rowM + kc * 32 + 16);
            v16h aw = pw[(size_t)kc * 32];
            acc = __builtin_amdgcn_wmma_f32_16x16x32_f16(false, aw, false, bm.v,
                                                         (short)0, acc, false, false);
        }
        int t = tile * FPB + nh * 16 + t16;
        if (t < NFRAMES) {
#pragma unroll
            for (int r = 0; r < 8; ++r) {
                int kbin = mt * 16 + (lo ? r : (r + 8));
                if (kbin < NBINS) {
                    out[((size_t)b * NBINS + kbin) * NFRAMES + t] =
                        __log10f(acc[r] + 1e-10f);
                }
            }
        }
    }
}

// ---------------------------------------------------------------------------
extern "C" void kernel_launch(void* const* d_in, const int* in_sizes, int n_in,
                              void* d_out, int out_size, void* d_ws, size_t ws_size,
                              hipStream_t stream) {
    (void)in_sizes; (void)n_in; (void)out_size; (void)ws_size;
    const float* audio = (const float*)d_in[0];
    float* out = (float*)d_out;

    // Workspace layout (all 32B-aligned): cos table | sin table | weights.
    char* ws = (char*)d_ws;
    _Float16* dcos = (_Float16*)(ws);
    _Float16* dsin = (_Float16*)(ws + (size_t)DFT_ELEMS * 2);
    _Float16* wswz = (_Float16*)(ws + (size_t)DFT_ELEMS * 4);

    cqt_dft_tables<<<(DFT_ELEMS + 255) / 256, 256, 0, stream>>>(dcos, dsin);
    cqt_weights<<<(W_ELEMS + 255) / 256, 256, 0, stream>>>(wswz);

    dim3 grid(T_TILES, 16);                    // 54 frame tiles x 16 batches
    size_t lds_bytes = (size_t)(FPB * SA_STRIDE + FPB * MAG_STRIDE) * sizeof(_Float16);
    cqt_main<<<grid, 256, lds_bytes, stream>>>(audio, dcos, dsin, wswz, out);
}
